// PromptEncoder_20109036880348
// MI455X (gfx1250) — matrix operand order repaired and checked by reading, not compiled, not run on hardware
//
#include <hip/hip_runtime.h>
#include <hip/hip_bf16.h>

typedef __attribute__((ext_vector_type(16))) __bf16 v16bf;
typedef __attribute__((ext_vector_type(8)))  __bf16 bf16x8;
typedef __attribute__((ext_vector_type(8)))  float  v8f;

union ABu { v16bf v; bf16x8 h[2]; };

#define BN_EPS 1e-5f

// ---------------------------------------------------------------------------
// Kernel 1: crop_and_resize  (B=64, C=3, 512x512 -> 224x224, f32 NCHW out)
// ---------------------------------------------------------------------------
__global__ void crop_resize_k(const float* __restrict__ img,
                              const float* __restrict__ box,
                              float* __restrict__ out)
{
    const int N = 64 * 3 * 224 * 224;
    int idx = blockIdx.x * blockDim.x + threadIdx.x;
    if (idx >= N) return;
    int ox = idx % 224;
    int oy = (idx / 224) % 224;
    int c  = (idx / (224 * 224)) % 3;
    int b  = idx / (3 * 224 * 224);

    const int W = 512, H = 512;
    int x1 = (int)rintf(box[b * 4 + 0]);
    int y1 = (int)rintf(box[b * 4 + 1]);
    int x2 = (int)rintf(box[b * 4 + 2]);
    int y2 = (int)rintf(box[b * 4 + 3]);
    x1 = min(max(x1, 0), W - 1);
    y1 = min(max(y1, 0), H - 1);
    x2 = min(max(x2, x1 + 1), W);
    y2 = min(max(y2, y1 + 1), H);

    int lenx = x2 - x1, leny = y2 - y1;

    float sx = fmaxf(((float)ox + 0.5f) * ((float)lenx / 224.0f) - 0.5f, 0.0f);
    int xlo = min((int)floorf(sx), lenx - 1);
    int xhi = min(xlo + 1, lenx - 1);
    float wx = sx - (float)xlo;
    int X0 = x1 + xlo, X1 = x1 + xhi;

    float sy = fmaxf(((float)oy + 0.5f) * ((float)leny / 224.0f) - 0.5f, 0.0f);
    int ylo = min((int)floorf(sy), leny - 1);
    int yhi = min(ylo + 1, leny - 1);
    float wy = sy - (float)ylo;
    int Y0 = y1 + ylo, Y1 = y1 + yhi;

    const float* p = img + ((long)(b * 3 + c)) * H * W;
    float v00 = p[Y0 * W + X0], v01 = p[Y0 * W + X1];
    float v10 = p[Y1 * W + X0], v11 = p[Y1 * W + X1];
    float top = v00 * (1.0f - wx) + v01 * wx;
    float bot = v10 * (1.0f - wx) + v11 * wx;
    out[idx] = top * (1.0f - wy) + bot * wy;
}

// ---------------------------------------------------------------------------
// Zero the 1-pixel spatial halo of a padded NHWC activation buffer.
// Perimeter pixels only: 2*Wp + 2*(Hp-2) per image.
// ---------------------------------------------------------------------------
__global__ void halo_zero_k(__bf16* __restrict__ act, int B, int Hp, int Wp, int C)
{
    int per = 2 * Wp + 2 * (Hp - 2);
    int N = B * per * C;
    int idx = blockIdx.x * blockDim.x + threadIdx.x;
    if (idx >= N) return;
    int c = idx % C;
    int p = (idx / C) % per;
    int b = idx / (C * per);
    int y, x;
    if (p < Wp)            { y = 0;      x = p; }
    else if (p < 2 * Wp)   { y = Hp - 1; x = p - Wp; }
    else { int q = p - 2 * Wp; y = 1 + (q >> 1); x = (q & 1) ? (Wp - 1) : 0; }
    act[((b * Hp + y) * Wp + x) * C + c] = (__bf16)0.0f;
}

// ---------------------------------------------------------------------------
// Kernel 2: conv1 direct (3 -> 48, stride 2, pad 1) + BN + ReLU.
// Output NHWC bf16 into the interior of a 114x114 zero-halo buffer,
// channel-padded 48 -> 64 so conv2's per-tap K-slab is a multiple of 32.
// ---------------------------------------------------------------------------
__global__ void conv1_k(const float* __restrict__ in, const float* __restrict__ w,
                        const float* __restrict__ g, const float* __restrict__ bb,
                        const float* __restrict__ m, const float* __restrict__ v,
                        __bf16* __restrict__ out)
{
    const int Ho = 112, Wo = 112, Hi = 224, Wi = 224, Ci = 3, Co = 48, Cp = 64;
    const int N = 64 * Ho * Wo * Cp;
    int idx = blockIdx.x * blockDim.x + threadIdx.x;
    if (idx >= N) return;
    int co = idx % Cp;
    int wo = (idx / Cp) % Wo;
    int ho = (idx / (Cp * Wo)) % Ho;
    int b  = idx / (Cp * Wo * Ho);

    int oidx = ((b * 114 + ho + 1) * 114 + wo + 1) * Cp + co;   // padded NHWC
    if (co >= Co) { out[oidx] = (__bf16)0.0f; return; }         // zero pad lanes

    float acc = 0.0f;
    for (int ci = 0; ci < Ci; ++ci)
        for (int kh = 0; kh < 3; ++kh) {
            int hi = ho * 2 - 1 + kh;
            if ((unsigned)hi >= (unsigned)Hi) continue;
            for (int kw = 0; kw < 3; ++kw) {
                int wi = wo * 2 - 1 + kw;
                if ((unsigned)wi >= (unsigned)Wi) continue;
                acc += in[((b * Ci + ci) * Hi + hi) * Wi + wi] *
                       w[((co * Ci + ci) * 3 + kh) * 3 + kw];
            }
        }
    float scale = g[co] / sqrtf(v[co] + BN_EPS);
    float y = (acc - m[co]) * scale + bb[co];
    out[oidx] = (__bf16)fmaxf(y, 0.0f);
}

// ---------------------------------------------------------------------------
// Weight repack: w[co][ci][taps] f32 -> wp[co][t*Cipad + ci] bf16 (zero pad).
// ---------------------------------------------------------------------------
__global__ void repack_w_k(const float* __restrict__ w, __bf16* __restrict__ wp,
                           int Ci, int Cipad, int taps, int Co)
{
    int N = Co * taps * Cipad;
    int idx = blockIdx.x * blockDim.x + threadIdx.x;
    if (idx >= N) return;
    int ci = idx % Cipad;
    int t  = (idx / Cipad) % taps;
    int co = idx / (Cipad * taps);
    wp[idx] = (ci < Ci) ? (__bf16)w[(co * Ci + ci) * taps + t] : (__bf16)0.0f;
}

// ---------------------------------------------------------------------------
// Kernel 3: 3x3 stride-2 conv as implicit GEMM on WMMA bf16, 2x2 blocked.
//   Input NHWC with 1-px zero halo -> every tap unconditionally in bounds:
//   the inner loop is pure b128 loads + v_wmma (no exec masking).
//   Each wave owns a 32(M)x32(N) tile: 4 independent f32 accumulators,
//   A/B fragments each reused twice, per K-step 8 loads : 4 WMMAs.
//   Fragment layouts (half = lane>>4, l15 = lane&15):
//     A: a[0..7] = K 8*half..+7, a[8..15] = K 16+8*half..+7 (rows = l15)
//     B: bf[0..7] = K 8*half..+7, bf[8..15] = K 16+8*half..+7 (cols = l15)
//     C/D: acc[i] -> row 8*half + i, col l15
// ---------------------------------------------------------------------------
__global__ void conv3x3_wmma_k(const __bf16* __restrict__ act,  // NHWC [B,Hi+2,Wi+2,Cipad]
                               const __bf16* __restrict__ wp,   // [Co][9*Cipad]
                               const float* __restrict__ bng, const float* __restrict__ bnb,
                               const float* __restrict__ bnm, const float* __restrict__ bnv,
                               __bf16* __restrict__ outB,       // NHWC [B,Hop,Wop,Co]
                               float* __restrict__ outF,        // optional f32 (pad==0 only)
                               int B, int Cipad, int Co, int Hi, int Wi, int Ho, int Wo,
                               int Hop, int Wop, int pad)
{
    const int lane = threadIdx.x & 31;
    const int wave = (blockIdx.x * blockDim.x + threadIdx.x) >> 5;
    const int Mtot = B * Ho * Wo;                 // multiple of 32 for all layers
    const int nSup = Co >> 5;
    if (wave >= (Mtot >> 5) * nSup) return;
    const int ms = wave / nSup, ns = wave % nSup;
    const int half = lane >> 4, l15 = lane & 15;
    const int Hp = Hi + 2, Wpd = Wi + 2;

    const int m0 = ms * 32 + l15;                 // A rows owned by this lane
    const int m1 = m0 + 16;
    const int b0 = m0 / (Ho * Wo), hw0 = m0 % (Ho * Wo);
    const int ho0 = hw0 / Wo, wo0 = hw0 % Wo;
    const int b1 = m1 / (Ho * Wo), hw1 = m1 % (Ho * Wo);
    const int ho1 = hw1 / Wo, wo1 = hw1 % Wo;

    const int n0 = ns * 32 + l15;                 // B cols owned by this lane
    const int n1 = n0 + 16;
    const int Ktot = 9 * Cipad;
    const __bf16* w0 = wp + n0 * Ktot;
    const __bf16* w1 = wp + n1 * Ktot;
    __builtin_prefetch(w0, 0, 3);
    __builtin_prefetch(w1, 0, 3);

    v8f acc00 = {}, acc01 = {}, acc10 = {}, acc11 = {};
    int tap = 0;
    for (int kh = 0; kh < 3; ++kh) {
        for (int kw = 0; kw < 3; ++kw, ++tap) {
            // halo coords: hi = ho*2 - 1 + kh + 1 = ho*2 + kh  (always in bounds)
            const __bf16* aA = act + ((b0 * Hp + ho0 * 2 + kh) * Wpd + wo0 * 2 + kw) * Cipad;
            const __bf16* aB = act + ((b1 * Hp + ho1 * 2 + kh) * Wpd + wo1 * 2 + kw) * Cipad;
            const __bf16* bb0 = w0 + tap * Cipad;
            const __bf16* bb1 = w1 + tap * Cipad;
            for (int c0 = 0; c0 < Cipad; c0 += 32) {
                ABu a0, a1, bv0, bv1;
                a0.h[0] = *(const bf16x8*)(aA + c0 + 8 * half);
                a0.h[1] = *(const bf16x8*)(aA + c0 + 16 + 8 * half);
                a1.h[0] = *(const bf16x8*)(aB + c0 + 8 * half);
                a1.h[1] = *(const bf16x8*)(aB + c0 + 16 + 8 * half);
                const __bf16* p0 = bb0 + c0 + 8 * half;
                const __bf16* p1 = bb1 + c0 + 8 * half;
                bv0.h[0] = *(const bf16x8*)(p0);
                bv0.h[1] = *(const bf16x8*)(p0 + 16);
                bv1.h[0] = *(const bf16x8*)(p1);
                bv1.h[1] = *(const bf16x8*)(p1 + 16);
                acc00 = __builtin_amdgcn_wmma_f32_16x16x32_bf16(false, a0.v, false, bv0.v,
                                                                (short)0, acc00, false, false);
                acc01 = __builtin_amdgcn_wmma_f32_16x16x32_bf16(false, a0.v, false, bv1.v,
                                                                (short)0, acc01, false, false);
                acc10 = __builtin_amdgcn_wmma_f32_16x16x32_bf16(false, a1.v, false, bv0.v,
                                                                (short)0, acc10, false, false);
                acc11 = __builtin_amdgcn_wmma_f32_16x16x32_bf16(false, a1.v, false, bv1.v,
                                                                (short)0, acc11, false, false);
            }
        }
    }

    float sc0 = bng[n0] / sqrtf(bnv[n0] + BN_EPS), be0 = bnb[n0], mn0 = bnm[n0];
    float sc1 = bng[n1] / sqrtf(bnv[n1] + BN_EPS), be1 = bnb[n1], mn1 = bnm[n1];

    auto store8 = [&](const v8f& acc, int mbase, int n, float sc, float be, float mn) {
#pragma unroll
        for (int i = 0; i < 8; ++i) {
            int mm = mbase + 8 * half + i;
            int ob = mm / (Ho * Wo);
            int oh = mm % (Ho * Wo);
            int oy = oh / Wo, ox = oh % Wo;
            float y = fmaxf((acc[i] - mn) * sc + be, 0.0f);
            int oidx = ((ob * Hop + oy + pad) * Wop + ox + pad) * Co + n;
            outB[oidx] = (__bf16)y;
            if (outF) outF[oidx] = y;     // only used with pad==0 (unpadded layout)
        }
    };
    store8(acc00, ms * 32,      n0, sc0, be0, mn0);
    store8(acc01, ms * 32,      n1, sc1, be1, mn1);
    store8(acc10, ms * 32 + 16, n0, sc0, be0, mn0);
    store8(acc11, ms * 32 + 16, n1, sc1, be1, mn1);
}

// ---------------------------------------------------------------------------
// Kernel 4: 1x1 local projection as WMMA GEMM, 2x2 blocked.
//   M = 64*196 tokens, N = 256, K = 256.  A = crop_feat NHWC bf16 (K-contig),
//   B = packed lp_w bf16.  out = local_tokens (B,196,256) in d_out.
// ---------------------------------------------------------------------------
__global__ void local_wmma_k(const __bf16* __restrict__ a4b, const __bf16* __restrict__ lpp,
                             const float* __restrict__ lpb, float* __restrict__ out)
{
    const int lane = threadIdx.x & 31;
    const int wave = (blockIdx.x * blockDim.x + threadIdx.x) >> 5;
    const int mSup = (64 * 196) >> 5;             // 392
    const int nSup = 256 >> 5;                    // 8
    if (wave >= mSup * nSup) return;
    const int ms = wave / nSup, ns = wave % nSup;
    const int half = lane >> 4, l15 = lane & 15;

    const int m0 = ms * 32 + l15, m1 = m0 + 16;
    const int n0 = ns * 32 + l15, n1 = n0 + 16;
    const __bf16* ar0 = a4b + m0 * 256;
    const __bf16* ar1 = a4b + m1 * 256;
    const __bf16* br0 = lpp + n0 * 256;
    const __bf16* br1 = lpp + n1 * 256;

    v8f acc00 = {}, acc01 = {}, acc10 = {}, acc11 = {};
    for (int k0 = 0; k0 < 256; k0 += 32) {
        ABu a0, a1, bv0, bv1;
        a0.h[0] = *(const bf16x8*)(ar0 + k0 + 8 * half);
        a0.h[1] = *(const bf16x8*)(ar0 + k0 + 16 + 8 * half);
        a1.h[0] = *(const bf16x8*)(ar1 + k0 + 8 * half);
        a1.h[1] = *(const bf16x8*)(ar1 + k0 + 16 + 8 * half);
        bv0.h[0] = *(const bf16x8*)(br0 + k0 + 8 * half);
        bv0.h[1] = *(const bf16x8*)(br0 + k0 + 16 + 8 * half);
        bv1.h[0] = *(const bf16x8*)(br1 + k0 + 8 * half);
        bv1.h[1] = *(const bf16x8*)(br1 + k0 + 16 + 8 * half);
        acc00 = __builtin_amdgcn_wmma_f32_16x16x32_bf16(false, a0.v, false, bv0.v,
                                                        (short)0, acc00, false, false);
        acc01 = __builtin_amdgcn_wmma_f32_16x16x32_bf16(false, a0.v, false, bv1.v,
                                                        (short)0, acc01, false, false);
        acc10 = __builtin_amdgcn_wmma_f32_16x16x32_bf16(false, a1.v, false, bv0.v,
                                                        (short)0, acc10, false, false);
        acc11 = __builtin_amdgcn_wmma_f32_16x16x32_bf16(false, a1.v, false, bv1.v,
                                                        (short)0, acc11, false, false);
    }
    float bi0 = lpb[n0], bi1 = lpb[n1];
#pragma unroll
    for (int i = 0; i < 8; ++i) {
        int r0 = ms * 32 + 8 * half + i;
        int r1 = r0 + 16;
        out[r0 * 256 + n0] = acc00[i] + bi0;
        out[r0 * 256 + n1] = acc01[i] + bi1;
        out[r1 * 256 + n0] = acc10[i] + bi0;
        out[r1 * 256 + n1] = acc11[i] + bi1;
    }
}

// ---------------------------------------------------------------------------
// Kernel 5: concat feature = [mean(crop_feat), mean(p5), box/512] -> (64,516)
// crop_feat is NHWC f32: a4f[(b*196 + t)*256 + c]
// ---------------------------------------------------------------------------
__global__ void feat516_k(const float* __restrict__ a4f, const float* __restrict__ p5,
                          const float* __restrict__ box, float* __restrict__ feat)
{
    int idx = blockIdx.x * blockDim.x + threadIdx.x;
    if (idx >= 64 * 516) return;
    int b = idx / 516, j = idx % 516;
    float v;
    if (j < 256) {
        const float* p = a4f + b * 196 * 256 + j;
        float s = 0.0f;
        for (int i = 0; i < 196; ++i) s += p[i * 256];
        v = s / 196.0f;
    } else if (j < 512) {
        const float* p = p5 + (b * 256 + (j - 256)) * 256;
        float s = 0.0f;
        for (int i = 0; i < 256; ++i) s += p[i];
        v = s / 256.0f;
    } else {
        v = box[b * 4 + (j - 512)] / 512.0f;      // W == H == 512
    }
    feat[idx] = v;
}

// ---------------------------------------------------------------------------
// Head kernels (tiny chained GEMMs, f32 VALU — launch-latency bound anyway)
// ---------------------------------------------------------------------------
__global__ void head_g_k(const float* __restrict__ feat, const float* __restrict__ gpw,
                         const float* __restrict__ gpb, const float* __restrict__ lemb,
                         const float* __restrict__ temb, const int* __restrict__ label,
                         const int* __restrict__ type, float* __restrict__ g)
{
    int idx = blockIdx.x * blockDim.x + threadIdx.x;
    if (idx >= 64 * 256) return;
    int b = idx >> 8, d = idx & 255;
    const float* f = feat + b * 516;
    const float* w = gpw + d * 516;
    float s = gpb[d];
    for (int k = 0; k < 516; ++k) s += f[k] * w[k];
    s += lemb[(label[b] + 1) * 256 + d] + temb[type[b] * 256 + d];
    g[idx] = s;
}

__global__ void head_h_k(const float* __restrict__ g, const float* __restrict__ w1,
                         const float* __restrict__ b1, float* __restrict__ h)
{
    int idx = blockIdx.x * blockDim.x + threadIdx.x;
    if (idx >= 64 * 512) return;
    int b = idx / 512, o = idx % 512;
    const float* gg = g + b * 256;
    const float* ww = w1 + o * 256;
    float s = b1[o];
    for (int k = 0; k < 256; ++k) s += gg[k] * ww[k];
    h[idx] = fmaxf(s, 0.0f);
}

__global__ void head_gp_k(const float* __restrict__ h, const float* __restrict__ w2,
                          const float* __restrict__ b2, float* __restrict__ out)
{
    int idx = blockIdx.x * blockDim.x + threadIdx.x;
    if (idx >= 64 * 256) return;
    int b = idx >> 8, d = idx & 255;
    const float* hh = h + b * 512;
    const float* ww = w2 + d * 512;
    float s = b2[d];
    for (int k = 0; k < 512; ++k) s += hh[k] * ww[k];
    out[idx] = s;   // global_prompt at d_out[0 : 16384)
}

__global__ void head_s_k(const float* __restrict__ gp,
                         const float* __restrict__ w3, const float* __restrict__ b3,
                         const float* __restrict__ w4, const float* __restrict__ b4,
                         const float* __restrict__ w5, const float* __restrict__ b5,
                         float* __restrict__ out)
{
    int idx = blockIdx.x * blockDim.x + threadIdx.x;
    if (idx >= 3 * 64 * 256) return;
    int which = idx / 16384;
    int r = idx % 16384;
    int b = r >> 8, d = r & 255;
    const float* w = (which == 0) ? w3 : (which == 1) ? w4 : w5;
    const float* bs = (which == 0) ? b3 : (which == 1) ? b4 : b5;
    const float* g = gp + b * 256;
    const float* ww = w + d * 256;
    float s = bs[d];
    for (int k = 0; k < 256; ++k) s += g[k] * ww[k];
    const long base = 16384L + 64L * 196 * 256;   // s3/s4/s5 follow local_tokens
    out[base + (long)which * 16384 + r] = s;
}

// ---------------------------------------------------------------------------
extern "C" void kernel_launch(void* const* d_in, const int* in_sizes, int n_in,
                              void* d_out, int out_size, void* d_ws, size_t ws_size,
                              hipStream_t stream)
{
    // Inputs in setup_inputs() dict order, params flattened leaf-by-leaf:
    const float* support_image = (const float*)d_in[0];
    const float* support_box   = (const float*)d_in[1];
    const float* support_p5    = (const float*)d_in[2];
    const int*   prompt_label  = (const int*)  d_in[3];
    const int*   prompt_type   = (const int*)  d_in[4];
    const float* conv_w[4] = { (const float*)d_in[5], (const float*)d_in[6],
                               (const float*)d_in[7], (const float*)d_in[8] };
    const float* bn_g[4] = { (const float*)d_in[9],  (const float*)d_in[10],
                             (const float*)d_in[11], (const float*)d_in[12] };
    const float* bn_b[4] = { (const float*)d_in[13], (const float*)d_in[14],
                             (const float*)d_in[15], (const float*)d_in[16] };
    const float* bn_m[4] = { (const float*)d_in[17], (const float*)d_in[18],
                             (const float*)d_in[19], (const float*)d_in[20] };
    const float* bn_v[4] = { (const float*)d_in[21], (const float*)d_in[22],
                             (const float*)d_in[23], (const float*)d_in[24] };
    const float* lp_w = (const float*)d_in[25];
    const float* lp_b = (const float*)d_in[26];
    const float* gp_w = (const float*)d_in[27];
    const float* gp_b = (const float*)d_in[28];
    const float* label_emb = (const float*)d_in[29];
    const float* type_emb  = (const float*)d_in[30];
    const float* sp_w3 = (const float*)d_in[31];
    const float* sp_b3 = (const float*)d_in[32];
    const float* sp_w4 = (const float*)d_in[33];
    const float* sp_b4 = (const float*)d_in[34];
    const float* sp_w5 = (const float*)d_in[35];
    const float* sp_b5 = (const float*)d_in[36];
    const float* r_w1 = (const float*)d_in[37];
    const float* r_b1 = (const float*)d_in[38];
    const float* r_w2 = (const float*)d_in[39];
    const float* r_b2 = (const float*)d_in[40];

    // Workspace layout (bytes, all chunks 256B-aligned)
    char* ws = (char*)d_ws;
    float*  crop = (float*)(ws);                  // 64*3*224*224 f32          38,535,168
    __bf16* a1p  = (__bf16*)(ws + 38535168UL);    // NHWC 64*114*114*64 halo  106,463,232
    __bf16* a2p  = (__bf16*)(ws + 144998400UL);   // NHWC 64*58*58*96  halo    41,336,832
    __bf16* a3p  = (__bf16*)(ws + 186335232UL);   // NHWC 64*30*30*192 halo    22,118,400
    float*  a4f  = (float*)(ws + 208453632UL);    // NHWC 64*196*256 f32       12,845,056
    __bf16* a4b  = (__bf16*)(ws + 221298688UL);   // NHWC 64*196*256 bf16       6,422,528
    __bf16* wp2  = (__bf16*)(ws + 227721216UL);   // 96  x 576  bf16              110,592
    __bf16* wp3  = (__bf16*)(ws + 227831808UL);   // 192 x 864  bf16              331,776
    __bf16* wp4  = (__bf16*)(ws + 228163584UL);   // 256 x 1728 bf16              884,736
    __bf16* lpp  = (__bf16*)(ws + 229048320UL);   // 256 x 256  bf16              131,072
    float*  feat = (float*)(ws + 229179392UL);    // 64*516
    float*  gbuf = (float*)(ws + 229311488UL);    // 64*256
    float*  hbuf = (float*)(ws + 229377024UL);    // 64*512

    float* out = (float*)d_out;

    // 0) weight repacks + halo zeroing (tiny, independent)
    repack_w_k<<<(96 * 9 * 64 + 255) / 256, 256, 0, stream>>>(conv_w[1], wp2, 48, 64, 9, 96);
    repack_w_k<<<(192 * 9 * 96 + 255) / 256, 256, 0, stream>>>(conv_w[2], wp3, 96, 96, 9, 192);
    repack_w_k<<<(256 * 9 * 192 + 255) / 256, 256, 0, stream>>>(conv_w[3], wp4, 192, 192, 9, 256);
    repack_w_k<<<(256 * 1 * 256 + 255) / 256, 256, 0, stream>>>(lp_w, lpp, 256, 256, 1, 256);
    halo_zero_k<<<(64 * 452 * 64 + 255) / 256, 256, 0, stream>>>(a1p, 64, 114, 114, 64);
    halo_zero_k<<<(64 * 228 * 96 + 255) / 256, 256, 0, stream>>>(a2p, 64, 58, 58, 96);
    halo_zero_k<<<(64 * 116 * 192 + 255) / 256, 256, 0, stream>>>(a3p, 64, 30, 30, 192);

    // 1) crop & resize
    {
        int n = 64 * 3 * 224 * 224;
        crop_resize_k<<<(n + 255) / 256, 256, 0, stream>>>(support_image, support_box, crop);
    }
    // 2) conv1 direct -> padded NHWC bf16 (64ch)
    {
        int n = 64 * 112 * 112 * 64;
        conv1_k<<<(n + 255) / 256, 256, 0, stream>>>(crop, conv_w[0], bn_g[0], bn_b[0],
                                                     bn_m[0], bn_v[0], a1p);
    }
    // 3) conv2..4 via WMMA implicit GEMM, 2x2 blocked (32x32 tile per wave)
    {
        int waves = (64 * 56 * 56 / 32) * (96 / 32);
        conv3x3_wmma_k<<<(waves + 7) / 8, 256, 0, stream>>>(a1p, wp2, bn_g[1], bn_b[1],
            bn_m[1], bn_v[1], a2p, nullptr, 64, 64, 96, 112, 112, 56, 56, 58, 58, 1);
    }
    {
        int waves = (64 * 28 * 28 / 32) * (192 / 32);
        conv3x3_wmma_k<<<(waves + 7) / 8, 256, 0, stream>>>(a2p, wp3, bn_g[2], bn_b[2],
            bn_m[2], bn_v[2], a3p, nullptr, 64, 96, 192, 56, 56, 28, 28, 30, 30, 1);
    }
    {
        int waves = (64 * 14 * 14 / 32) * (256 / 32);
        conv3x3_wmma_k<<<(waves + 7) / 8, 256, 0, stream>>>(a3p, wp4, bn_g[3], bn_b[3],
            bn_m[3], bn_v[3], a4b, a4f, 64, 192, 256, 28, 28, 14, 14, 14, 14, 0);
    }
    // 4) local tokens (1x1 conv GEMM) straight into d_out[16384 : )
    {
        int waves = (64 * 196 / 32) * (256 / 32);
        local_wmma_k<<<(waves + 7) / 8, 256, 0, stream>>>(a4b, lpp, lp_b, out + 16384);
    }
    // 5) concat features
    {
        int n = 64 * 516;
        feat516_k<<<(n + 255) / 256, 256, 0, stream>>>(a4f, support_p5, support_box, feat);
    }
    // 6) head chain
    head_g_k<<<(64 * 256 + 255) / 256, 256, 0, stream>>>(feat, gp_w, gp_b, label_emb,
                                                         type_emb, prompt_label, prompt_type, gbuf);
    head_h_k<<<(64 * 512 + 255) / 256, 256, 0, stream>>>(gbuf, r_w1, r_b1, hbuf);
    head_gp_k<<<(64 * 256 + 255) / 256, 256, 0, stream>>>(hbuf, r_w2, r_b2, out);
    head_s_k<<<(3 * 64 * 256 + 255) / 256, 256, 0, stream>>>(out, sp_w3, sp_b3, sp_w4, sp_b4,
                                                             sp_w5, sp_b5, out);
    (void)in_sizes; (void)n_in; (void)out_size; (void)ws_size;
}
